// GNNNodeClassifier_78915729097325
// MI455X (gfx1250) — matrix operand order repaired and checked by reading, not compiled, not run on hardware
//
#include <hip/hip_runtime.h>
#include <hip/hip_bf16.h>
#include <math.h>

typedef __attribute__((ext_vector_type(16))) __bf16 v16bf;
typedef __attribute__((ext_vector_type(8)))  __bf16 v8bf;
typedef __attribute__((ext_vector_type(2)))  __bf16 v2bf;
typedef __attribute__((ext_vector_type(8)))  float  v8f;

#define BN_EPS 1e-5f

// ---------------------------------------------------------------------------
// Fold eval-mode BatchNorm into the linear layer and transpose to bf16:
//   s[k] = g[k] * rsqrt(v[k]+eps);  t[k] = b[k] - m[k]*s[k]
//   Wt[n*K + k] = bf16(s[k] * W[k*Nout + n])   (column-major bf16)
//   biasOut[n]  = bias[n] + sum_k t[k]*W[k*Nout + n]
// g==nullptr -> identity BN (output projection). grid = Nout, 128 threads.
// ---------------------------------------------------------------------------
__global__ void fold_kernel(const float* __restrict__ g, const float* __restrict__ b,
                            const float* __restrict__ m, const float* __restrict__ v,
                            const float* __restrict__ W, const float* __restrict__ bias,
                            __bf16* __restrict__ Wt, float* __restrict__ biasOut,
                            int K, int Nout)
{
    __shared__ float red[128];
    const int n = blockIdx.x;
    const int t = threadIdx.x;
    float partial = 0.f;
    for (int k = t; k < K; k += 128) {
        float s = 1.f, tt = 0.f;
        if (g != nullptr) {
            s  = g[k] * rsqrtf(v[k] + BN_EPS);
            tt = b[k] - m[k] * s;
        }
        float w = W[(size_t)k * Nout + n];
        Wt[(size_t)n * K + k] = (__bf16)(s * w);
        partial += tt * w;
    }
    red[t] = partial;
    __syncthreads();
    for (int off = 64; off > 0; off >>= 1) {
        if (t < off) red[t] += red[t + off];
        __syncthreads();
    }
    if (t == 0) biasOut[n] = bias[n] + red[0];
}

// ---------------------------------------------------------------------------
// WMMA FFN:  Cout[M, Nout] = act( A[M, K] @ W' + bias' )
// A fp32 comes from two pointers (k<128 -> A0, k>=128 -> A1, which handles
// concat([x, agg]) unmaterialized); if divCnt != nullptr the A1 half is
// scaled by 1/max(divCnt[row],1) during staging (fused scatter-mean divide).
//
// The 16-row x K A-panel is converted to bf16 ONCE per block into LDS
// (row stride K+8 elements -> 4-dword skew per row, conflict-free
// ds_load_b128 fragment reads), shared by all Nout/16 waves.
//
// Fragment layouts used here (wave32, contiguous k order, same for A and B):
//   A (16x32 bf16): lane row = lane&15, kb = kk + (lane>>4)*16,
//                   elems 0..7 -> K=kb..kb+7, elems 8..15 -> K=kb+8..kb+15
//   B (32x16 bf16): lane col = lane&15, elem e -> K = kk + (lane>>4)*16 + e
//   C (16x16 f32) : elem r -> row = r + 8*(lane>>4), col = lane&15
// blockDim.x = 32*(Nout/16), gridDim.x = ceil(M/16).
// ---------------------------------------------------------------------------
__global__ void ffn_wmma_kernel(const float* __restrict__ A0, int lda0,
                                const float* __restrict__ A1, int lda1,
                                const float* __restrict__ divCnt,
                                const __bf16* __restrict__ Wt,
                                const float* __restrict__ biasF,
                                float* __restrict__ Cout, int ldc,
                                const int* __restrict__ gatherIdx,
                                int M, int K, int applyGelu)
{
    __shared__ __bf16 ldsA[16 * 264];          // max K=256, stride 264 elems
    const int SE = K + 8;                       // LDS row stride (elements)

    const int wave = threadIdx.x >> 5;
    const int lane = threadIdx.x & 31;
    const int mi   = lane & 15;
    const int hi   = lane >> 4;
    const int rowbase = blockIdx.x << 4;
    const int colbase = wave << 4;

    // ---- cooperative stage: fp32 (x2 sources, optional 1/cnt) -> bf16 LDS ----
    const int K2 = K >> 1;                      // bf16 pairs per row
    for (int idx = threadIdx.x; idx < 16 * K2; idx += blockDim.x) {
        const int r  = idx / K2;
        const int k  = (idx - r * K2) * 2;      // even; pair never crosses 128
        int row = rowbase + r;
        if (row >= M) row = M - 1;              // M%16==0 in practice
        const int gr = gatherIdx ? gatherIdx[row] : row;
        float s0, s1;
        if (k < 128) {
            const float* p = A0 + (size_t)gr * lda0 + k;
            s0 = p[0]; s1 = p[1];
        } else {
            const float* p = A1 + (size_t)gr * lda1 + (k - 128);
            float inv = 1.f;
            if (divCnt) inv = 1.f / fmaxf(divCnt[gr], 1.f);
            s0 = p[0] * inv; s1 = p[1] * inv;
        }
        *(v2bf*)(&ldsA[r * SE + k]) = (v2bf){(__bf16)s0, (__bf16)s1};
    }
    __syncthreads();

    // ---- WMMA main loop: A from LDS, B (weights) from global/L0 ----
    const __bf16* wcol = Wt + (size_t)(colbase + mi) * K;
    v8f acc = (v8f){0.f,0.f,0.f,0.f,0.f,0.f,0.f,0.f};
    for (int kk = 0; kk < K; kk += 32) {
        const int kb0 = kk + hi * 16;
        const v8bf alo = *(const v8bf*)(&ldsA[mi * SE + kb0]);
        const v8bf ahi = *(const v8bf*)(&ldsA[mi * SE + kb0 + 8]);
        v16bf a;
        #pragma unroll
        for (int e = 0; e < 8; ++e) { a[e] = alo[e]; a[e + 8] = ahi[e]; }
        v16bf bfrag = *(const v16bf*)(wcol + kk + hi * 16);   // 32B contiguous
        acc = __builtin_amdgcn_wmma_f32_16x16x32_bf16(
                  false, a, false, bfrag, (short)0, acc, false, false);
    }

    // ---- epilogue: bias + exact GELU, store fp32 ----
    const int col = colbase + mi;
    const float bb = biasF[col];
    #pragma unroll
    for (int r = 0; r < 8; ++r) {
        const int row = rowbase + r + hi * 8;
        if (row < M) {
            float x = acc[r] + bb;
            if (applyGelu) x = 0.5f * x * (1.f + erff(x * 0.70710678118654752f));
            Cout[(size_t)row * ldc + col] = x;
        }
    }
}

// ---------------------------------------------------------------------------
__global__ void zero_kernel(float* __restrict__ p, long n)
{
    long i = (long)blockIdx.x * blockDim.x + threadIdx.x;
    if (i < n) p[i] = 0.f;
}

// Degree of each src node (identical for both convs -> computed once).
__global__ void degree_kernel(const int* __restrict__ src, float* __restrict__ cnt, int E)
{
    int i = blockIdx.x * blockDim.x + threadIdx.x;
    if (i < E) atomicAdd(cnt + src[i], 1.0f);
}

// One wave per edge: agg[src] += y[dst]  (128 f32 atomics; L2-resident table).
__global__ void scatter_kernel(const int* __restrict__ src, const int* __restrict__ dst,
                               const float* __restrict__ y,
                               float* __restrict__ agg, int E)
{
    const int e    = blockIdx.x * (blockDim.x >> 5) + (threadIdx.x >> 5);
    const int lane = threadIdx.x & 31;
    if (e >= E) return;
    const int s = src[e], d = dst[e];
    const float* yr = y   + (size_t)d * 128;
    float*       ar = agg + (size_t)s * 128;
    #pragma unroll
    for (int i = 0; i < 4; ++i)
        atomicAdd(ar + lane + 32 * i, yr[lane + 32 * i]);
}

// ---------------------------------------------------------------------------
extern "C" void kernel_launch(void* const* d_in, const int* in_sizes, int n_in,
                              void* d_out, int out_size, void* d_ws, size_t ws_size,
                              hipStream_t stream) {
    (void)n_in; (void)out_size; (void)ws_size;

    const float* node_features = (const float*)d_in[0];
    const int*   edge_src      = (const int*)d_in[1];
    const int*   edge_dst      = (const int*)d_in[2];
    const int*   node_idx      = (const int*)d_in[3];

    const int N = in_sizes[3];
    const int E = in_sizes[1];
    const int F = in_sizes[0] / N;      // 256
    const int H = 128, C = 32;

    const float* P[42];
    for (int i = 0; i < 42; ++i) P[i] = (const float*)d_in[i];

    // ---- carve workspace (256B-aligned chunks) ----
    size_t off = 0;
    auto carve = [&](size_t bytes) -> void* {
        void* p = (char*)d_ws + off;
        off += (bytes + 255) & ~(size_t)255;
        return p;
    };
    float*  bufA = (float*)carve((size_t)N * H * 4);
    float*  bufB = (float*)carve((size_t)N * H * 4);
    float*  bufC = (float*)carve((size_t)N * H * 4);   // agg accumulator
    float*  cnt  = (float*)carve((size_t)N * 4);
    __bf16* WtPre  = (__bf16*)carve((size_t)F * H * 2);     float* bPre  = (float*)carve(H * 4);
    __bf16* WtC1p  = (__bf16*)carve((size_t)H * H * 2);     float* bC1p  = (float*)carve(H * 4);
    __bf16* WtC1u  = (__bf16*)carve((size_t)2 * H * H * 2); float* bC1u  = (float*)carve(H * 4);
    __bf16* WtC2p  = (__bf16*)carve((size_t)H * H * 2);     float* bC2p  = (float*)carve(H * 4);
    __bf16* WtC2u  = (__bf16*)carve((size_t)2 * H * H * 2); float* bC2u  = (float*)carve(H * 4);
    __bf16* WtPost = (__bf16*)carve((size_t)H * H * 2);     float* bPost = (float*)carve(H * 4);
    __bf16* WtOut  = (__bf16*)carve((size_t)H * C * 2);     float* bOut  = (float*)carve(C * 4);

    // ---- fold BN into weights, transpose to bf16 ----
    fold_kernel<<<H, 128, 0, stream>>>(P[ 4], P[ 5], P[ 6], P[ 7], P[ 8], P[ 9], WtPre,  bPre,  F,     H);
    fold_kernel<<<H, 128, 0, stream>>>(P[10], P[11], P[12], P[13], P[14], P[15], WtC1p,  bC1p,  H,     H);
    fold_kernel<<<H, 128, 0, stream>>>(P[16], P[17], P[18], P[19], P[20], P[21], WtC1u,  bC1u,  2 * H, H);
    fold_kernel<<<H, 128, 0, stream>>>(P[22], P[23], P[24], P[25], P[26], P[27], WtC2p,  bC2p,  H,     H);
    fold_kernel<<<H, 128, 0, stream>>>(P[28], P[29], P[30], P[31], P[32], P[33], WtC2u,  bC2u,  2 * H, H);
    fold_kernel<<<H, 128, 0, stream>>>(P[34], P[35], P[36], P[37], P[38], P[39], WtPost, bPost, H,     H);
    fold_kernel<<<C, 128, 0, stream>>>(nullptr, nullptr, nullptr, nullptr, P[40], P[41], WtOut, bOut, H, C);

    const int  rowBlocks  = (N + 15) / 16;            // 50000/16 = 3125 exact
    const long aggElems   = (long)N * H;
    const int  zeroBlocks = (int)((aggElems + 255) / 256);
    const int  edgeBlocks = (E + 7) / 8;

    // degree once (edge_src identical for both convs)
    zero_kernel<<<(N + 255) / 256, 256, 0, stream>>>(cnt, N);
    degree_kernel<<<(E + 255) / 256, 256, 0, stream>>>(edge_src, cnt, E);

    // x = preFFN(node_features)                                 -> bufA
    ffn_wmma_kernel<<<rowBlocks, 256, 0, stream>>>(node_features, F, node_features, F, nullptr,
                                                   WtPre, bPre, bufA, H, nullptr, N, F, 1);
    // ---- conv1: per-node prepare FFN, scatter-add, fused mean in update ----
    ffn_wmma_kernel<<<rowBlocks, 256, 0, stream>>>(bufA, H, bufA, H, nullptr,
                                                   WtC1p, bC1p, bufB, H, nullptr, N, H, 1);
    zero_kernel<<<zeroBlocks, 256, 0, stream>>>(bufC, aggElems);
    scatter_kernel<<<edgeBlocks, 256, 0, stream>>>(edge_src, edge_dst, bufB, bufC, E);
    // x1 = c1uFFN(concat(x, agg/deg))                           -> bufB
    ffn_wmma_kernel<<<rowBlocks, 256, 0, stream>>>(bufA, H, bufC, H, cnt,
                                                   WtC1u, bC1u, bufB, H, nullptr, N, 2 * H, 1);
    // ---- conv2 ----
    ffn_wmma_kernel<<<rowBlocks, 256, 0, stream>>>(bufB, H, bufB, H, nullptr,
                                                   WtC2p, bC2p, bufA, H, nullptr, N, H, 1);
    zero_kernel<<<zeroBlocks, 256, 0, stream>>>(bufC, aggElems);
    scatter_kernel<<<edgeBlocks, 256, 0, stream>>>(edge_src, edge_dst, bufA, bufC, E);
    // x2 = c2uFFN(concat(x1, agg/deg))                          -> bufA
    ffn_wmma_kernel<<<rowBlocks, 256, 0, stream>>>(bufB, H, bufC, H, cnt,
                                                   WtC2u, bC2u, bufA, H, nullptr, N, 2 * H, 1);
    // post FFN                                                  -> bufB
    ffn_wmma_kernel<<<rowBlocks, 256, 0, stream>>>(bufA, H, bufA, H, nullptr,
                                                   WtPost, bPost, bufB, H, nullptr, N, H, 1);
    // out = emb[node_idx] @ out_W + out_b  (Nout=32 -> 2 waves, no GELU)
    ffn_wmma_kernel<<<rowBlocks, 64, 0, stream>>>(bufB, H, bufB, H, nullptr,
                                                  WtOut, bOut, (float*)d_out, C,
                                                  node_idx, N, H, 0);
}
